// Linformer_43894565765677
// MI455X (gfx1250) — compile-verified, matrix operand order checked by reading, not run on hardware
//
#include <hip/hip_runtime.h>
#include <math.h>

// ---------------------------------------------------------------------------
// Linformer forward for MI455X (gfx1250): bf16 WMMA GEMMs with double-buffered
// async (memory -> LDS) staging + fused epilogues.
// ---------------------------------------------------------------------------

typedef __bf16 bf16;
typedef __bf16 v16bf __attribute__((ext_vector_type(16)));
typedef __bf16 v8bf  __attribute__((ext_vector_type(8)));
typedef float  v8f   __attribute__((ext_vector_type(8)));

#define FLAG_OUT_BF16 1
#define FLAG_TRANS    2
#define FLAG_GELU     4

#define BB  2
#define NN  4096
#define CC  1024
#define HH  16
#define DH  64
#define KK  256
#define FFD 4096
#define BK  32
#define BKP 40   // padded LDS row (80B): conflict-free b128 fragment loads

// CDNA5 async DMA: global -> LDS, 16B per lane, tracked by ASYNCcnt.
__device__ inline void async_cp16(unsigned lds_byte_off, const bf16* g) {
  asm volatile("global_load_async_to_lds_b128 %0, %1, off"
               :: "v"(lds_byte_off), "v"(g)
               : "memory");
}
__device__ inline void wait_async0() {
  asm volatile("s_wait_asynccnt 0x0" ::: "memory");
}
// Low 32 bits of a flat LDS address are the workgroup-relative LDS offset.
__device__ inline unsigned lds_off(const void* p) {
  return (unsigned)(unsigned long long)p;
}

// Load one WMMA operand fragment (16x32 bf16, A or B^T layout) from an LDS row.
// Lane L: m/n = L%16, khalf = L/16 ; holds K = khalf*8..+7 and 16+khalf*8..+7.
__device__ inline v16bf load_frag(const bf16* __restrict__ row, int kh) {
  const v8bf* p = (const v8bf*)row;
  v8bf lo = p[kh];
  v8bf hi = p[2 + kh];
  v16bf r;
#pragma unroll
  for (int i = 0; i < 8; ++i) { r[i] = lo[i]; r[8 + i] = hi[i]; }
  return r;
}

__device__ inline float gelu_exact(float v) {
  return 0.5f * v * (1.0f + erff(v * 0.70710678118654752f));
}

// C[m,n] = alpha * sum_k A[m,k] * Bt[n,k]  (+ bias) (+gelu), batched over Z.
// A rows are K-contiguous (async-staged). Bt rows are K-contiguous when
// ldb_k==1 (async-staged), else gathered synchronously (E-projection).
template <int BM, int BN, int WR, int WC>
__global__ __launch_bounds__(WR * WC * 32)
void gemm_bf16_wmma(const bf16* __restrict__ A, long long lda_m,
                    long long sAb, long long sAh,
                    const bf16* __restrict__ Bt, long long ldb_n, long long ldb_k,
                    long long sBb, long long sBh,
                    void* __restrict__ Cp, long long ldc,
                    long long sCb, long long sCh,
                    const float* __restrict__ bias, int bias_mode, float alpha,
                    int Kd, int Hb, int flags) {
  constexpr int THREADS = WR * WC * 32;
  __shared__ bf16 As[2][BM * BKP];
  __shared__ bf16 Bs[2][BN * BKP];

  const int tid = threadIdx.x;
  const int z = blockIdx.z;
  const int zb = z / Hb, zh = z % Hb;
  const int bm0 = blockIdx.y * BM;
  const int bn0 = blockIdx.x * BN;

  const bf16* Ab = A + (size_t)zb * sAb + (size_t)zh * sAh + (size_t)bm0 * lda_m;
  const bf16* Bb = Bt + (size_t)zb * sBb + (size_t)zh * sBh;

  const int lane = tid & 31, wave = tid >> 5;
  const int wr = (wave / WC) * 64, wc = (wave % WC) * 64;
  const int m16 = lane & 15, kh = lane >> 4;

  v8f acc[4][4];
#pragma unroll
  for (int i = 0; i < 4; ++i)
#pragma unroll
    for (int j = 0; j < 4; ++j) acc[i][j] = (v8f){0.f, 0.f, 0.f, 0.f, 0.f, 0.f, 0.f, 0.f};

  // Stage one 32-deep K-tile into LDS buffer `buf` (async where contiguous).
  auto stage = [&](int buf, int k0) {
    for (int r = tid; r < BM; r += THREADS) {
      const bf16* src = Ab + (size_t)r * lda_m + k0;
      unsigned dst = lds_off(&As[buf][r * BKP]);
#pragma unroll
      for (int c = 0; c < 4; ++c) async_cp16(dst + c * 16, src + c * 8);
    }
    if (ldb_k == 1) {
      for (int r = tid; r < BN; r += THREADS) {
        const bf16* src = Bb + (size_t)(bn0 + r) * ldb_n + k0;
        unsigned dst = lds_off(&Bs[buf][r * BKP]);
#pragma unroll
        for (int c = 0; c < 4; ++c) async_cp16(dst + c * 16, src + c * 8);
      }
    } else {
      // strided-K gather; consecutive lanes hit consecutive addresses (ldb_n==1)
      for (int idx = tid; idx < BN * BK; idx += THREADS) {
        int n = idx % BN, kk = idx / BN;
        Bs[buf][n * BKP + kk] =
            Bb[(size_t)(bn0 + n) * ldb_n + (size_t)(k0 + kk) * ldb_k];
      }
    }
  };

  const int nk = Kd / BK;
  stage(0, 0);
  for (int t = 0; t < nk; ++t) {
    wait_async0();       // my async writes for tile t landed in LDS
    __syncthreads();     // everyone's tile-t writes visible
    if (t + 1 < nk) stage((t + 1) & 1, (t + 1) * BK);  // prefetch next tile

    const bf16* Abuf = As[t & 1];
    const bf16* Bbuf = Bs[t & 1];
    v16bf af[4], bfr[4];
#pragma unroll
    for (int i = 0; i < 4; ++i) af[i] = load_frag(&Abuf[(wr + i * 16 + m16) * BKP], kh);
#pragma unroll
    for (int j = 0; j < 4; ++j) bfr[j] = load_frag(&Bbuf[(wc + j * 16 + m16) * BKP], kh);
#pragma unroll
    for (int i = 0; i < 4; ++i)
#pragma unroll
      for (int j = 0; j < 4; ++j)
        acc[i][j] = __builtin_amdgcn_wmma_f32_16x16x32_bf16(
            false, af[i], false, bfr[j], (short)0, acc[i][j], false, false);
    __syncthreads();     // all reads of buf (t&1) done before it is re-staged
  }

  // ---- epilogue: alpha, bias, gelu, optional transpose, bf16/f32 store ----
  const size_t cofs = (size_t)zb * sCb + (size_t)zh * sCh;
#pragma unroll
  for (int i = 0; i < 4; ++i)
#pragma unroll
    for (int j = 0; j < 4; ++j)
#pragma unroll
      for (int r = 0; r < 8; ++r) {
        int gm = bm0 + wr + i * 16 + kh * 8 + r;
        int gn = bn0 + wc + j * 16 + m16;
        float v = acc[i][j][r] * alpha;
        if (bias_mode == 1) v += bias[gn];
        else if (bias_mode == 2) v += bias[gm];
        if (flags & FLAG_GELU) v = gelu_exact(v);
        size_t idx = cofs + ((flags & FLAG_TRANS) ? (size_t)gn * ldc + gm
                                                  : (size_t)gm * ldc + gn);
        if (flags & FLAG_OUT_BF16) ((bf16*)Cp)[idx] = (bf16)v;
        else                       ((float*)Cp)[idx] = v;
      }
}

// Row softmax over K=256 (one wave per row; lane covers 8 strided elements).
__global__ __launch_bounds__(256)
void softmax_k(const bf16* __restrict__ s, bf16* __restrict__ p, size_t nrows) {
  const int lane = threadIdx.x & 31, wv = threadIdx.x >> 5;
  const size_t row = (size_t)blockIdx.x * 8 + wv;
  if (row >= nrows) return;
  const bf16* sr = s + row * KK;
  bf16* pr = p + row * KK;
  float v[8];
#pragma unroll
  for (int i = 0; i < 8; ++i) v[i] = (float)sr[lane + 32 * i];
  float m = v[0];
#pragma unroll
  for (int i = 1; i < 8; ++i) m = fmaxf(m, v[i]);
  for (int o = 16; o; o >>= 1) m = fmaxf(m, __shfl_xor(m, o, 32));
  float sum = 0.f;
#pragma unroll
  for (int i = 0; i < 8; ++i) { v[i] = __expf(v[i] - m); sum += v[i]; }
  for (int o = 16; o; o >>= 1) sum += __shfl_xor(sum, o, 32);
  float inv = 1.0f / sum;
#pragma unroll
  for (int i = 0; i < 8; ++i) pr[lane + 32 * i] = (bf16)(v[i] * inv);
}

// Fused residual-add + LayerNorm over C=1024; writes fp32 x and bf16 x.
__global__ __launch_bounds__(256)
void add_ln_k(const float* __restrict__ x, const float* __restrict__ y,
              const float* __restrict__ g, const float* __restrict__ bta,
              float* __restrict__ xo, bf16* __restrict__ xbo) {
  __shared__ float s1[8], s2[8];
  const size_t row = blockIdx.x;
  const int tid = threadIdx.x;
  const float* xr = x + row * CC;
  const float* yr = y + row * CC;
  float r[4]; float sum = 0.f, sq = 0.f;
#pragma unroll
  for (int i = 0; i < 4; ++i) {
    int c = tid + 256 * i;
    float t = xr[c] + yr[c];
    r[i] = t; sum += t; sq += t * t;
  }
  for (int o = 16; o; o >>= 1) { sum += __shfl_xor(sum, o, 32); sq += __shfl_xor(sq, o, 32); }
  const int lane = tid & 31, wv = tid >> 5;
  if (lane == 0) { s1[wv] = sum; s2[wv] = sq; }
  __syncthreads();
  float ts = 0.f, tq = 0.f;
#pragma unroll
  for (int i = 0; i < 8; ++i) { ts += s1[i]; tq += s2[i]; }
  const float mean = ts * (1.0f / CC);
  const float var = tq * (1.0f / CC) - mean * mean;
  const float inv = rsqrtf(var + 1e-5f);
#pragma unroll
  for (int i = 0; i < 4; ++i) {
    int c = tid + 256 * i;
    float o = (r[i] - mean) * inv * g[c] + bta[c];
    xo[row * CC + c] = o;
    xbo[row * CC + c] = (bf16)o;
  }
}

// ---- fp32 -> bf16 packing kernels --------------------------------------
__global__ void cvt_bf16_k(const float* __restrict__ src, bf16* __restrict__ dst, size_t n) {
  for (size_t i = (size_t)blockIdx.x * blockDim.x + threadIdx.x; i < n;
       i += (size_t)gridDim.x * blockDim.x)
    dst[i] = (bf16)src[i];
}

// WqkvT[j, c] for j = sel*1024 + h*64 + d  <-  W{q,k,v}[h, c, d]
__global__ void pack_qkv_k(const float* __restrict__ Wq, const float* __restrict__ Wk,
                           const float* __restrict__ Wv, bf16* __restrict__ dst) {
  const size_t n = (size_t)3072 * 1024;
  for (size_t i = (size_t)blockIdx.x * blockDim.x + threadIdx.x; i < n;
       i += (size_t)gridDim.x * blockDim.x) {
    int j = (int)(i >> 10), c = (int)(i & 1023);
    int sel = j >> 10, jm = j & 1023, h = jm >> 6, d = jm & 63;
    const float* W = (sel == 0) ? Wq : (sel == 1) ? Wk : Wv;
    dst[i] = (bf16)W[((size_t)h * CC + c) * DH + d];
  }
}

// dst[c, r] = src[r, c]   (R x Cc  ->  Cc x R), fp32 -> bf16
__global__ void transpose_cvt_k(const float* __restrict__ src, bf16* __restrict__ dst,
                                int R, int Cc) {
  const size_t n = (size_t)R * Cc;
  for (size_t i = (size_t)blockIdx.x * blockDim.x + threadIdx.x; i < n;
       i += (size_t)gridDim.x * blockDim.x) {
    int r = (int)(i / Cc), c = (int)(i % Cc);
    dst[(size_t)c * R + r] = (bf16)src[i];
  }
}

extern "C" void kernel_launch(void* const* d_in, const int* in_sizes, int n_in,
                              void* d_out, int out_size, void* d_ws, size_t ws_size,
                              hipStream_t stream) {
  (void)in_sizes; (void)n_in; (void)out_size; (void)ws_size;
  const float* x_in = (const float*)d_in[0];
  const float* Wq  = (const float*)d_in[1];
  const float* Wk  = (const float*)d_in[2];
  const float* Wv  = (const float*)d_in[3];
  const float* Wo  = (const float*)d_in[4];
  const float* bo  = (const float*)d_in[5];
  const float* Ew  = (const float*)d_in[6];
  const float* Eb  = (const float*)d_in[7];
  const float* W1  = (const float*)d_in[8];
  const float* b1  = (const float*)d_in[9];
  const float* W2  = (const float*)d_in[10];
  const float* b2  = (const float*)d_in[11];
  const float* lng = (const float*)d_in[12];
  const float* lnb = (const float*)d_in[13];

  char* base = (char*)d_ws;
  size_t off = 0;
  auto take = [&](size_t bytes) -> void* {
    void* p = base + off;
    off += (bytes + 255) & ~(size_t)255;
    return p;
  };

  const size_t MR = (size_t)BB * NN;           // 8192 rows
  bf16* xbf   = (bf16*)take(MR * CC * 2);
  float* xf   = (float*)take(MR * CC * 4);
  bf16* WqkvT = (bf16*)take((size_t)3072 * 1024 * 2);
  bf16* WoT   = (bf16*)take((size_t)1024 * 1024 * 2);
  bf16* W1T   = (bf16*)take((size_t)FFD * CC * 2);
  bf16* W2T   = (bf16*)take((size_t)CC * FFD * 2);
  bf16* Ewb   = (bf16*)take((size_t)KK * NN * 2);
  bf16* qkv   = (bf16*)take(MR * 3072 * 2);
  bf16* kp    = (bf16*)take((size_t)BB * HH * KK * DH * 2);
  bf16* vpT   = (bf16*)take((size_t)BB * HH * DH * KK * 2);
  bf16* sc    = (bf16*)take((size_t)BB * HH * NN * KK * 2);
  bf16* pbuf  = (bf16*)take((size_t)BB * HH * NN * KK * 2);
  bf16* obuf  = (bf16*)take(MR * CC * 2);
  float* attn = (float*)take(MR * CC * 4);
  bf16* hbuf  = sc;  // FF hidden reuses the dead scores buffer (same size)

  cvt_bf16_k<<<8192, 256, 0, stream>>>(x_in, xbf, MR * CC);
  cvt_bf16_k<<<4096, 256, 0, stream>>>(Ew, Ewb, (size_t)KK * NN);

  for (int l = 0; l < 2; ++l) {
    const size_t wofs = (size_t)l * HH * CC * DH;
    pack_qkv_k<<<12288, 256, 0, stream>>>(Wq + wofs, Wk + wofs, Wv + wofs, WqkvT);
    transpose_cvt_k<<<4096, 256, 0, stream>>>(Wo + (size_t)l * CC * CC, WoT, 1024, 1024);
    transpose_cvt_k<<<8192, 256, 0, stream>>>(W1 + (size_t)l * CC * FFD, W1T, CC, FFD);
    transpose_cvt_k<<<8192, 256, 0, stream>>>(W2 + (size_t)l * FFD * CC, W2T, FFD, CC);

    // 1) fused QKV projection: [8192,1024] x [1024,3072] -> qkv bf16
    gemm_bf16_wmma<128, 128, 2, 2><<<dim3(3072 / 128, MR / 128, 1), 128, 0, stream>>>(
        xbf, CC, 0, 0, WqkvT, CC, 1, 0, 0, qkv, 3072, 0, 0,
        nullptr, 0, 1.0f, CC, 1, FLAG_OUT_BF16);

    // 2) kp[b,h] = Ew (256x4096) x K[b,:,h,:] (+Eb per-row)
    gemm_bf16_wmma<128, 64, 2, 1><<<dim3(1, KK / 128, BB * HH), 64, 0, stream>>>(
        Ewb, NN, 0, 0,
        qkv + 1024, 1, 3072, (long long)NN * 3072, DH,
        kp, DH, (long long)HH * KK * DH, (long long)KK * DH,
        Eb, 2, 1.0f, NN, HH, FLAG_OUT_BF16);

    // 3) vp[b,h], stored transposed (Dh x K) so the o-GEMM gets contiguous B^T
    gemm_bf16_wmma<128, 64, 2, 1><<<dim3(1, KK / 128, BB * HH), 64, 0, stream>>>(
        Ewb, NN, 0, 0,
        qkv + 2048, 1, 3072, (long long)NN * 3072, DH,
        vpT, KK, (long long)HH * DH * KK, (long long)DH * KK,
        Eb, 2, 1.0f, NN, HH, FLAG_OUT_BF16 | FLAG_TRANS);

    // 4) scores[b,h] = (q · kp^T) * 1/sqrt(Dh)   (Kd = Dh = 64)
    gemm_bf16_wmma<128, 128, 2, 2><<<dim3(KK / 128, NN / 128, BB * HH), 128, 0, stream>>>(
        qkv, 3072, (long long)NN * 3072, DH,
        kp, DH, 1, (long long)HH * KK * DH, (long long)KK * DH,
        sc, KK, (long long)HH * NN * KK, (long long)NN * KK,
        nullptr, 0, 0.125f, DH, HH, FLAG_OUT_BF16);

    // 5) softmax over K=256
    softmax_k<<<(BB * HH * NN) / 8, 256, 0, stream>>>(sc, pbuf, (size_t)BB * HH * NN);

    // 6) o[b,h] = P (4096x256) x Vp (256x64) -> packed heads [B,N,H*Dh]
    gemm_bf16_wmma<128, 64, 2, 1><<<dim3(1, NN / 128, BB * HH), 64, 0, stream>>>(
        pbuf, KK, (long long)HH * NN * KK, (long long)NN * KK,
        vpT, KK, 1, (long long)HH * DH * KK, (long long)DH * KK,
        obuf, CC, (long long)NN * CC, DH,
        nullptr, 0, 1.0f, KK, HH, FLAG_OUT_BF16);

    // 7) output projection (+bo), fp32 out
    gemm_bf16_wmma<128, 128, 2, 2><<<dim3(CC / 128, MR / 128, 1), 128, 0, stream>>>(
        obuf, CC, 0, 0, WoT, CC, 1, 0, 0, attn, CC, 0, 0,
        bo + (size_t)l * CC, 1, 1.0f, CC, 1, 0);

    // 8) residual + LN (attn)
    add_ln_k<<<MR, 256, 0, stream>>>((l == 0) ? x_in : xf, attn,
                                     lng + (size_t)(l * 2 + 0) * CC,
                                     lnb + (size_t)(l * 2 + 0) * CC, xf, xbf);

    // 9) FF1 (+b1, exact gelu), bf16 out
    gemm_bf16_wmma<128, 128, 2, 2><<<dim3(FFD / 128, MR / 128, 1), 128, 0, stream>>>(
        xbf, CC, 0, 0, W1T, CC, 1, 0, 0, hbuf, FFD, 0, 0,
        b1 + (size_t)l * FFD, 1, 1.0f, CC, 1, FLAG_OUT_BF16 | FLAG_GELU);

    // 10) FF2 (+b2), fp32 out
    gemm_bf16_wmma<128, 128, 2, 2><<<dim3(CC / 128, MR / 128, 1), 128, 0, stream>>>(
        hbuf, FFD, 0, 0, W2T, FFD, 1, 0, 0, attn, CC, 0, 0,
        b2 + (size_t)l * CC, 1, 1.0f, FFD, 1, 0);

    // 11) residual + LN (ff); final layer writes d_out
    float* xout = (l == 1) ? (float*)d_out : xf;
    add_ln_k<<<MR, 256, 0, stream>>>(xf, attn,
                                     lng + (size_t)(l * 2 + 1) * CC,
                                     lnb + (size_t)(l * 2 + 1) * CC, xout, xbf);
  }
}
